// Net_69672959476237
// MI455X (gfx1250) — matrix operand order repaired and checked by reading, not compiled, not run on hardware
//
#include <hip/hip_runtime.h>
#include <hip/hip_bf16.h>

#define EPS 1e-5f

typedef __bf16 bf16_t;
typedef __bf16 v16bf __attribute__((ext_vector_type(16)));
typedef __bf16 v8bf  __attribute__((ext_vector_type(8)));
typedef float  v8f   __attribute__((ext_vector_type(8)));

constexpr int Nn = 16384;  // nodes
constexpr int Ff = 512;    // input features
constexpr int Hh = 512;    // hidden
constexpr int Cc = 128;    // out features

constexpr int BK   = 64;    // K-tile per stage (2 WMMA k-halves)
constexpr int ROWB = 144;   // 128B tile row + 16B pad (bank-conflict relief, 16B aligned)

// ---------------- elementwise / conversion kernels ----------------

__global__ void bn1_cvt_kernel(const float* __restrict__ x,
                               const float* __restrict__ g1, const float* __restrict__ beta1,
                               const float* __restrict__ rm1, const float* __restrict__ rv1,
                               bf16_t* __restrict__ xbT) {
  int idx = blockIdx.x * blockDim.x + threadIdx.x;
  if (idx >= Nn * Ff) return;
  int n = idx / Ff, f = idx % Ff;
  float v = (x[idx] - rm1[f]) * rsqrtf(rv1[f] + EPS) * g1[f] + beta1[f];
  xbT[(size_t)f * Nn + n] = (bf16_t)v;
}

__global__ void cvt_a_kernel(const float* __restrict__ a, bf16_t* __restrict__ ab) {
  size_t idx = ((size_t)blockIdx.x * blockDim.x + threadIdx.x) * 8;
  float4 p0 = *(const float4*)(a + idx);
  float4 p1 = *(const float4*)(a + idx + 4);
  v8bf o;
  o[0] = (bf16_t)p0.x; o[1] = (bf16_t)p0.y; o[2] = (bf16_t)p0.z; o[3] = (bf16_t)p0.w;
  o[4] = (bf16_t)p1.x; o[5] = (bf16_t)p1.y; o[6] = (bf16_t)p1.z; o[7] = (bf16_t)p1.w;
  *(v8bf*)(ab + idx) = o;
}

__global__ void cvt_T_kernel(const float* __restrict__ w, bf16_t* __restrict__ wT,
                             int rows, int cols) {
  int idx = blockIdx.x * blockDim.x + threadIdx.x;
  if (idx >= rows * cols) return;
  int r = idx / cols, c = idx % cols;
  wT[(size_t)c * rows + r] = (bf16_t)w[idx];
}

// ---------------- WMMA GEMM with async global->LDS double buffering ----------------
// C[M,Ncols] = A[M,K](bf16 row-major, lda==K) @ Bt[Ncols,K](bf16, K contiguous per column)
// Block: 256 threads = 8 waves (4M x 2N); block tile 128x128; wave tile 32x64; BK=64.
// Each stage: 32 KB (A 128x128B + B 128x128B) via 8 async b128 DMAs per wave.
template <int EPI>
__global__ __launch_bounds__(256) void gemm_kernel(
    const bf16_t* __restrict__ A, const bf16_t* __restrict__ Bt,
    int M, int Ncols, int K,
    bf16_t* __restrict__ outb, float* __restrict__ outf, bf16_t* __restrict__ outbT,
    const float* __restrict__ bias,
    const float* __restrict__ g2, const float* __restrict__ beta2,
    const float* __restrict__ rm2, const float* __restrict__ rv2) {
  __shared__ __align__(16) char smA[2][128 * ROWB];
  __shared__ __align__(16) char smB[2][128 * ROWB];

  const int tid   = threadIdx.x;
  const int lane  = tid & 31;
  const int wave  = tid >> 5;
  const int waveM = wave >> 1;   // 0..3
  const int waveN = wave & 1;    // 0..1
  const int m0blk = blockIdx.y * 128;
  const int n0blk = blockIdx.x * 128;
  const int l15 = lane & 15;
  const bool hi = lane >= 16;
  const int aoff = hi ? 8 : 0;    // A frag: hi half-wave holds K+8..15 / K+24..31
  const int boff = hi ? 8 : 0;    // B frag: hi half-wave holds K+8..15 / K+24..31

  // -------- staging assignment (async DMA): 8 lanes x 16B per 128B row --------
  const bool stageB = wave >= 4;
  const int  sw     = stageB ? (wave - 4) : wave;     // 0..3
  const int  srow0  = sw * 32 + (lane >> 3);          // + i*4, i=0..7
  const int  sbyte  = (lane & 7) * 16;                // 16B per lane per op
  const bf16_t* sbase = stageB ? Bt : A;
  const int  growb  = stageB ? n0blk : m0blk;

  auto stage = [&](int buf, int k0) {
    char* sm = stageB ? smB[buf] : smA[buf];
#pragma unroll
    for (int i = 0; i < 8; ++i) {
      const int row = srow0 + i * 4;
      const unsigned long long g =
          (unsigned long long)(uintptr_t)(sbase + (size_t)(growb + row) * K + k0) + sbyte;
      const unsigned l = (unsigned)(uintptr_t)(sm + row * ROWB + sbyte);
      asm volatile("global_load_async_to_lds_b128 %0, %1, off"
                   :: "v"(l), "v"(g) : "memory");
    }
  };

  v8f acc[2][4];
#pragma unroll
  for (int i = 0; i < 2; ++i)
#pragma unroll
    for (int j = 0; j < 4; ++j)
#pragma unroll
      for (int r = 0; r < 8; ++r) acc[i][j][r] = 0.0f;

  const int nk = K / BK;
  stage(0, 0);

  for (int kt = 0; kt < nk; ++kt) {
    const int cur = kt & 1;
    if (kt + 1 < nk) {
      stage(cur ^ 1, (kt + 1) * BK);
      asm volatile("s_wait_asynccnt 0x8" ::: "memory");   // stage kt complete (in-order)
    } else {
      asm volatile("s_wait_asynccnt 0x0" ::: "memory");
    }
    __syncthreads();   // all waves' DMA for stage kt visible in LDS

#pragma unroll
    for (int h = 0; h < 2; ++h) {   // two WMMA k-halves per stage
      v16bf af[2], bfr[4];
#pragma unroll
      for (int i = 0; i < 2; ++i) {
        const char* pa =
            &smA[cur][(waveM * 32 + 16 * i + l15) * ROWB + h * 64 + aoff * 2];
        v8bf lo = *(const v8bf*)pa;
        v8bf hh = *(const v8bf*)(pa + 32);
        af[i] = __builtin_shufflevector(lo, hh,
                 0, 1, 2, 3, 4, 5, 6, 7, 8, 9, 10, 11, 12, 13, 14, 15);
      }
#pragma unroll
      for (int j = 0; j < 4; ++j) {
        const char* pb =
            &smB[cur][(waveN * 64 + 16 * j + l15) * ROWB + h * 64 + boff * 2];
        v8bf lo = *(const v8bf*)pb;
        v8bf hh = *(const v8bf*)(pb + 32);
        bfr[j] = __builtin_shufflevector(lo, hh,
                  0, 1, 2, 3, 4, 5, 6, 7, 8, 9, 10, 11, 12, 13, 14, 15);
      }
#pragma unroll
      for (int i = 0; i < 2; ++i)
#pragma unroll
        for (int j = 0; j < 4; ++j)
          acc[i][j] = __builtin_amdgcn_wmma_f32_16x16x32_bf16(
              false, af[i], false, bfr[j], (short)0, acc[i][j], false, false);
    }

    __syncthreads();   // reads of buf[cur] done before stage kt+2 overwrites it
  }

  // ---------------- epilogue ----------------
#pragma unroll
  for (int j = 0; j < 4; ++j) {
    const int n = n0blk + waveN * 64 + 16 * j + l15;
    float sc = 0.f, sh = 0.f, bb = 0.f;
    if (EPI == 1) {
      float s = g2[n] * rsqrtf(rv2[n] + EPS);
      sc = s; sh = beta2[n] - rm2[n] * s; bb = bias[n];
    } else if (EPI == 2) {
      bb = bias[n];
    }
#pragma unroll
    for (int i = 0; i < 2; ++i) {
      const int mb = m0blk + waveM * 32 + 16 * i + (hi ? 8 : 0);
#pragma unroll
      for (int r = 0; r < 8; ++r) {
        const int m = mb + r;
        float v = acc[i][j][r];
        if (EPI == 0) {
          outb[(size_t)m * Ncols + n] = (bf16_t)v;
        } else if (EPI == 1) {
          float y = v + bb;
          y = fmaxf(y, 0.0f);
          y = y * sc + sh;
          outf[(size_t)m * Ncols + n] = y;            // x2 (fp32 output)
          outbT[(size_t)n * M + m] = (bf16_t)y;       // x2^T bf16 (B-operand for next GEMM)
        } else {
          outf[(size_t)m * Ncols + n] = v + bb;       // x1 (fp32 output)
        }
      }
    }
  }
}

// ---------------- launcher ----------------
extern "C" void kernel_launch(void* const* d_in, const int* in_sizes, int n_in,
                              void* d_out, int out_size, void* d_ws, size_t ws_size,
                              hipStream_t stream) {
  (void)in_sizes; (void)n_in; (void)out_size; (void)ws_size;
  const float* x     = (const float*)d_in[0];
  const float* a     = (const float*)d_in[1];
  const float* w1    = (const float*)d_in[2];
  const float* b1    = (const float*)d_in[3];
  const float* w2    = (const float*)d_in[4];
  const float* b2    = (const float*)d_in[5];
  const float* g1    = (const float*)d_in[6];
  const float* beta1 = (const float*)d_in[7];
  const float* rm1   = (const float*)d_in[8];
  const float* rv1   = (const float*)d_in[9];
  const float* g2    = (const float*)d_in[10];
  const float* beta2 = (const float*)d_in[11];
  const float* rm2   = (const float*)d_in[12];
  const float* rv2   = (const float*)d_in[13];

  float* out_x1 = (float*)d_out;                 // [N, C]
  float* out_x2 = out_x1 + (size_t)Nn * Cc;      // [N, H]

  char* ws = (char*)d_ws;
  bf16_t* ab   = (bf16_t*)ws; ws += (size_t)Nn * Nn * 2;  // 512 MB
  bf16_t* xbT  = (bf16_t*)ws; ws += (size_t)Ff * Nn * 2;  // 16 MB
  bf16_t* t1b  = (bf16_t*)ws; ws += (size_t)Nn * Hh * 2;  // 16 MB
  bf16_t* x2bT = (bf16_t*)ws; ws += (size_t)Hh * Nn * 2;  // 16 MB
  bf16_t* t2b  = (bf16_t*)ws; ws += (size_t)Nn * Hh * 2;  // 16 MB
  bf16_t* w1bT = (bf16_t*)ws; ws += (size_t)Ff * Hh * 2;
  bf16_t* w2bT = (bf16_t*)ws; ws += (size_t)Hh * Cc * 2;

  bn1_cvt_kernel<<<(Nn * Ff) / 256, 256, 0, stream>>>(x, g1, beta1, rm1, rv1, xbT);
  cvt_a_kernel<<<(int)(((size_t)Nn * Nn) / (256 * 8)), 256, 0, stream>>>(a, ab);
  cvt_T_kernel<<<(Ff * Hh) / 256, 256, 0, stream>>>(w1, w1bT, Ff, Hh);
  cvt_T_kernel<<<(Hh * Cc) / 256, 256, 0, stream>>>(w2, w2bT, Hh, Cc);

  dim3 blk(256);
  // t1 = a @ BN1(x)
  gemm_kernel<0><<<dim3(Hh / 128, Nn / 128), blk, 0, stream>>>(
      ab, xbT, Nn, Hh, Nn, t1b, nullptr, nullptr, nullptr, nullptr, nullptr, nullptr, nullptr);
  // x2 = BN2(relu(t1 @ w1 + b1))
  gemm_kernel<1><<<dim3(Hh / 128, Nn / 128), blk, 0, stream>>>(
      t1b, w1bT, Nn, Hh, Ff, nullptr, out_x2, x2bT, b1, g2, beta2, rm2, rv2);
  // t2 = a @ x2
  gemm_kernel<0><<<dim3(Hh / 128, Nn / 128), blk, 0, stream>>>(
      ab, x2bT, Nn, Hh, Nn, t2b, nullptr, nullptr, nullptr, nullptr, nullptr, nullptr, nullptr);
  // x1 = t2 @ w2 + b2
  gemm_kernel<2><<<dim3(Cc / 128, Nn / 128), blk, 0, stream>>>(
      t2b, w2bT, Nn, Cc, Hh, nullptr, out_x1, nullptr, b2, nullptr, nullptr, nullptr, nullptr);
}
